// PromptDurationMemoryEncoder_15814069584479
// MI455X (gfx1250) — matrix operand order repaired and checked by reading, not compiled, not run on hardware
//
#include <hip/hip_runtime.h>
#include <hip/hip_bf16.h>
#include <math.h>

#define BB 16
#define TT 4096
#define DD 512
#define MM 256
#define KCONV 3
#define BT (BB*TT)
#define BTD ((size_t)BB*TT*DD)
#define LN_EPS 1e-5f

typedef __attribute__((ext_vector_type(16))) __bf16 v16bf;
typedef __attribute__((ext_vector_type(8)))  float  v8f;
typedef __attribute__((ext_vector_type(4)))  unsigned int u32x4;

union HalfPack { u32x4 q[2]; v16bf v; };

__device__ __forceinline__ __bf16 f2bf(float f) {
  unsigned u = __builtin_bit_cast(unsigned, f);
  unsigned r = u + 0x7FFFu + ((u >> 16) & 1u);
  unsigned short s = (unsigned short)(r >> 16);
  return __builtin_bit_cast(__bf16, s);
}
__device__ __forceinline__ float bf2f(__bf16 h) {
  unsigned short s = __builtin_bit_cast(unsigned short, h);
  unsigned u = ((unsigned)s) << 16;
  return __builtin_bit_cast(float, u);
}
__device__ __forceinline__ float wsum(float v) {
  for (int o = 16; o > 0; o >>= 1) v += __shfl_xor(v, o, 32);
  return v;
}
__device__ __forceinline__ float wmaxr(float v) {
  for (int o = 16; o > 0; o >>= 1) v = fmaxf(v, __shfl_xor(v, o, 32));
  return v;
}
__device__ __forceinline__ void zero8(v8f& a) {
#pragma unroll
  for (int r = 0; r < 8; r++) a[r] = 0.0f;
}

// ---------- elementwise prologue ----------
__global__ void k_mask_logdur(const float* __restrict__ dur, const float* __restrict__ pm,
                              float* __restrict__ mask, float* __restrict__ logdur) {
  int i = blockIdx.x * blockDim.x + threadIdx.x;
  if (i >= BT) return;
  float m = fminf(fmaxf(pm[i], 0.f), 1.f);
  mask[i] = m;
  logdur[i] = logf(fmaxf(dur[i], 1e-4f)) * m;
}

// ---------- masked lower-median via bitonic sort in LDS ----------
__global__ void k_median(const float* __restrict__ logdur, const float* __restrict__ mask,
                         float* __restrict__ grate, float* __restrict__ support) {
  __shared__ float s[TT];
  __shared__ int csh[256];
  int b = blockIdx.x, tid = threadIdx.x;
  int cnt = 0;
  for (int i = tid; i < TT; i += 256) {
    float mk = mask[b * TT + i];
    bool valid = mk > 0.5f;
    s[i] = valid ? logdur[b * TT + i] : __builtin_inff();
    cnt += valid ? 1 : 0;
  }
  csh[tid] = cnt;
  __syncthreads();
  for (int k = 2; k <= TT; k <<= 1) {
    for (int j = k >> 1; j > 0; j >>= 1) {
      for (int i = tid; i < TT; i += 256) {
        int l = i ^ j;
        if (l > i) {
          bool up = ((i & k) == 0);
          float a = s[i], c = s[l];
          if ((a > c) == up) { s[i] = c; s[l] = a; }
        }
      }
      __syncthreads();
    }
  }
  for (int st = 128; st > 0; st >>= 1) {
    if (tid < st) csh[tid] += csh[tid + st];
    __syncthreads();
  }
  if (tid == 0) {
    int n = csh[0];
    int idx = n > 0 ? (n - 1) / 2 : 0;
    grate[b] = n > 0 ? s[idx] : 0.0f;
    support[b] = fmaxf((float)n, 1.0f);
  }
}

__global__ void k_residual(const float* __restrict__ logdur, const float* __restrict__ mask,
                           const float* __restrict__ grate, float* __restrict__ rres) {
  int i = blockIdx.x * blockDim.x + threadIdx.x;
  if (i >= BT) return;
  int b = i / TT;
  rres[i] = (logdur[i] - grate[b]) * mask[i];
}

__global__ void k_zero(float* __restrict__ p, int n) {
  int i = blockIdx.x * blockDim.x + threadIdx.x;
  if (i < n) p[i] = 0.0f;
}

// ---------- weight prep: fp32 [O,I,K] -> bf16 [O][kk*512+i] ----------
__global__ void k_prep_convw(const float* __restrict__ w, __bf16* __restrict__ wt) {
  int i = blockIdx.x * blockDim.x + threadIdx.x;
  if (i >= DD * DD * KCONV) return;
  int o = i / (DD * KCONV);
  int rem = i % (DD * KCONV);
  int kk = rem / DD;
  int ci = rem % DD;
  wt[i] = f2bf(w[(o * DD + ci) * KCONV + kk]);
}

__global__ void k_prep_bf16(const float* __restrict__ in, __bf16* __restrict__ out, int n) {
  int i = blockIdx.x * blockDim.x + threadIdx.x;
  if (i < n) out[i] = f2bf(in[i]);
}

// ---------- h0 = emb[u] + logdur*proj_w[:,0] + proj_b ----------
__global__ void k_hidden0(const int* __restrict__ units, const float* __restrict__ logdur,
                          const float* __restrict__ emb, const float* __restrict__ proj_w,
                          const float* __restrict__ proj_b, __bf16* __restrict__ h0) {
  size_t i = (size_t)blockIdx.x * blockDim.x + threadIdx.x;
  if (i >= BTD) return;
  int d = (int)(i & (DD - 1));
  size_t bt = i >> 9;
  int u = units[bt];
  float v = emb[(size_t)u * DD + d] + logdur[bt] * proj_w[d * 4] + proj_b[d];
  h0[i] = f2bf(v);
}

// ---------- causal conv (K=3) as WMMA GEMM, 2x2 register-blocked ----------
// Each wave computes a 32(t) x 32(n) output block: 4 accumulators, 4 WMMAs per
// (A0,A1,B0,B1) load group -> 2x the arithmetic intensity of 1x1 tiling.
__global__ void __launch_bounds__(256)
k_conv_wmma(const __bf16* __restrict__ hin, const __bf16* __restrict__ wt,
            const float* __restrict__ bias, __bf16* __restrict__ hout) {
  int lane = threadIdx.x & 31, wave = threadIdx.x >> 5;
  int b = blockIdx.z;
  int ncol0 = blockIdx.y * 32;
  int tblk = blockIdx.x * 8 + wave;   // 32-row strip index
  int m = lane & 15, h = lane >> 4;
  int trow0 = tblk * 32 + m;
  int trow1 = trow0 + 16;
  const __bf16* wcol0 = wt + (size_t)(ncol0 + m) * 1536;
  const __bf16* wcol1 = wt + (size_t)(ncol0 + 16 + m) * 1536;
  v8f a00, a01, a10, a11;
  zero8(a00); zero8(a01); zero8(a10); zero8(a11);
  for (int c = 0; c < 48; ++c) {
    int kk = c >> 4;
    int i0 = (c & 15) << 5;
    int s0 = trow0 + kk - 2;   // causal left pad (only strip 0 can underflow)
    int s1 = trow1 + kk - 2;
    HalfPack A0, A1, B0, B1;
    if (s0 >= 0) {
      const __bf16* ar = hin + (((size_t)b * TT + s0) << 9) + i0;
      A0.q[0] = *(const u32x4*)(ar + h * 8);
      A0.q[1] = *(const u32x4*)(ar + 16 + h * 8);
    } else {
#pragma unroll
      for (int e = 0; e < 4; e++) { A0.q[0][e] = 0u; A0.q[1][e] = 0u; }
    }
    {
      const __bf16* ar = hin + (((size_t)b * TT + s1) << 9) + i0;
      A1.q[0] = *(const u32x4*)(ar + h * 8);
      A1.q[1] = *(const u32x4*)(ar + 16 + h * 8);
    }
    const __bf16* br0 = wcol0 + c * 32 + h * 8;
    const __bf16* br1 = wcol1 + c * 32 + h * 8;
    B0.q[0] = *(const u32x4*)(br0);
    B0.q[1] = *(const u32x4*)(br0 + 16);
    B1.q[0] = *(const u32x4*)(br1);
    B1.q[1] = *(const u32x4*)(br1 + 16);
    a00 = __builtin_amdgcn_wmma_f32_16x16x32_bf16(false, A0.v, false, B0.v, (short)0, a00, false, false);
    a01 = __builtin_amdgcn_wmma_f32_16x16x32_bf16(false, A0.v, false, B1.v, (short)0, a01, false, false);
    a10 = __builtin_amdgcn_wmma_f32_16x16x32_bf16(false, A1.v, false, B0.v, (short)0, a10, false, false);
    a11 = __builtin_amdgcn_wmma_f32_16x16x32_bf16(false, A1.v, false, B1.v, (short)0, a11, false, false);
  }
#pragma unroll
  for (int ti = 0; ti < 2; ti++) {
#pragma unroll
    for (int ni = 0; ni < 2; ni++) {
      v8f acc = ti == 0 ? (ni == 0 ? a00 : a01) : (ni == 0 ? a10 : a11);
      int n = ncol0 + ni * 16 + m;
      float bs = bias[n];
#pragma unroll
      for (int r = 0; r < 8; r++) {
        int tt = tblk * 32 + ti * 16 + r + h * 8;
        float x = acc[r] + bs;
        float g = 0.5f * x * (1.0f + erff(x * 0.70710678118654752f));
        hout[(((size_t)b * TT + tt) << 9) + n] = f2bf(g);
      }
    }
  }
}

// ---------- layernorm * mask (fp32 math, bf16 in/out), wave per row ----------
__global__ void __launch_bounds__(256)
k_layernorm(const __bf16* __restrict__ hin, const float* __restrict__ g,
            const float* __restrict__ be, const float* __restrict__ mask,
            __bf16* __restrict__ hout) {
  int lane = threadIdx.x & 31, wave = threadIdx.x >> 5;
  size_t row = (size_t)blockIdx.x * 8 + wave;
  const __bf16* x = hin + (row << 9);
  float vals[16];
  float s = 0.0f;
#pragma unroll
  for (int k = 0; k < 16; k++) { float v = bf2f(x[lane + k * 32]); vals[k] = v; s += v; }
  s = wsum(s);
  float mu = s * (1.0f / 512.0f);
  float q = 0.0f;
#pragma unroll
  for (int k = 0; k < 16; k++) { float d = vals[k] - mu; q += d * d; }
  q = wsum(q);
  float inv = rsqrtf(q * (1.0f / 512.0f) + LN_EPS);
  float mk = mask[row];
  __bf16* o = hout + (row << 9);
#pragma unroll
  for (int k = 0; k < 16; k++) {
    int d = lane + k * 32;
    o[d] = f2bf(((vals[k] - mu) * inv * g[d] + be[d]) * mk);
  }
}

// ---------- score = hidden @ role_key^T / sqrt(D), WMMA 2x2 blocked ----------
__global__ void __launch_bounds__(256)
k_score_wmma(const __bf16* __restrict__ hln, const __bf16* __restrict__ rk,
             float* __restrict__ score) {
  int lane = threadIdx.x & 31, wave = threadIdx.x >> 5;
  int b = blockIdx.z;
  int ncol0 = blockIdx.y * 32;
  int tblk = blockIdx.x * 8 + wave;
  int m = lane & 15, h = lane >> 4;
  int trow0 = tblk * 32 + m;
  const __bf16* ar0 = hln + (((size_t)b * TT + trow0) << 9);
  const __bf16* ar1 = hln + (((size_t)b * TT + trow0 + 16) << 9);
  const __bf16* wcol0 = rk + (size_t)(ncol0 + m) * DD;
  const __bf16* wcol1 = rk + (size_t)(ncol0 + 16 + m) * DD;
  v8f a00, a01, a10, a11;
  zero8(a00); zero8(a01); zero8(a10); zero8(a11);
  for (int c = 0; c < 16; ++c) {
    int i0 = c << 5;
    HalfPack A0, A1, B0, B1;
    A0.q[0] = *(const u32x4*)(ar0 + i0 + h * 8);
    A0.q[1] = *(const u32x4*)(ar0 + i0 + 16 + h * 8);
    A1.q[0] = *(const u32x4*)(ar1 + i0 + h * 8);
    A1.q[1] = *(const u32x4*)(ar1 + i0 + 16 + h * 8);
    B0.q[0] = *(const u32x4*)(wcol0 + i0 + h * 8);
    B0.q[1] = *(const u32x4*)(wcol0 + i0 + 16 + h * 8);
    B1.q[0] = *(const u32x4*)(wcol1 + i0 + h * 8);
    B1.q[1] = *(const u32x4*)(wcol1 + i0 + 16 + h * 8);
    a00 = __builtin_amdgcn_wmma_f32_16x16x32_bf16(false, A0.v, false, B0.v, (short)0, a00, false, false);
    a01 = __builtin_amdgcn_wmma_f32_16x16x32_bf16(false, A0.v, false, B1.v, (short)0, a01, false, false);
    a10 = __builtin_amdgcn_wmma_f32_16x16x32_bf16(false, A1.v, false, B0.v, (short)0, a10, false, false);
    a11 = __builtin_amdgcn_wmma_f32_16x16x32_bf16(false, A1.v, false, B1.v, (short)0, a11, false, false);
  }
  const float sc = 0.044194173824159216f;  // 1/sqrt(512)
#pragma unroll
  for (int ti = 0; ti < 2; ti++) {
#pragma unroll
    for (int ni = 0; ni < 2; ni++) {
      v8f acc = ti == 0 ? (ni == 0 ? a00 : a01) : (ni == 0 ? a10 : a11);
      int n = ncol0 + ni * 16 + m;
#pragma unroll
      for (int r = 0; r < 8; r++) {
        int tt = tblk * 32 + ti * 16 + r + h * 8;
        score[(((size_t)b * TT + tt) << 8) + n] = acc[r] * sc;
      }
    }
  }
}

// ---------- softmax over M in place; attn = mask>0 ? softmax : 0 ----------
__global__ void __launch_bounds__(256)
k_softmax(float* __restrict__ attn, const float* __restrict__ mask) {
  int lane = threadIdx.x & 31, wave = threadIdx.x >> 5;
  size_t row = (size_t)blockIdx.x * 8 + wave;
  float* p = attn + (row << 8);
  float mk = mask[row];
  if (mk > 0.0f) {
    float v[8];
    float mx = -__builtin_inff();
#pragma unroll
    for (int k = 0; k < 8; k++) { v[k] = p[lane + k * 32]; mx = fmaxf(mx, v[k]); }
    mx = wmaxr(mx);
    float s = 0.0f;
#pragma unroll
    for (int k = 0; k < 8; k++) { v[k] = expf(v[k] - mx); s += v[k]; }
    s = wsum(s);
    float inv = 1.0f / s;
#pragma unroll
    for (int k = 0; k < 8; k++) p[lane + k * 32] = v[k] * inv;
  } else {
#pragma unroll
    for (int k = 0; k < 8; k++) p[lane + k * 32] = 0.0f;
  }
}

// ---------- deterministic two-stage reductions over T ----------
__global__ void k_part1(const float* __restrict__ attn, const float* __restrict__ rres,
                        float* __restrict__ pcov, float* __restrict__ prv) {
  int b = blockIdx.x, sl = blockIdx.y, m = threadIdx.x;
  int t0 = sl * (TT / 16);
  const float* ap = attn + (((size_t)b * TT + t0) << 8) + m;
  const float* rp = rres + b * TT + t0;
  float c = 0.0f, rv = 0.0f;
  for (int t = 0; t < TT / 16; t++) {
    float a = ap[(size_t)t << 8];
    c += a;
    rv += a * rp[t];
  }
  pcov[(b * 16 + sl) * MM + m] = c;
  prv[(b * 16 + sl) * MM + m] = rv;
}

__global__ void k_fin1(const float* __restrict__ pcov, const float* __restrict__ prv,
                       const float* __restrict__ support, float* __restrict__ cov,
                       float* __restrict__ rval, float* __restrict__ rcov) {
  int b = blockIdx.x, m = threadIdx.x;
  float c = 0.0f, rv = 0.0f;
  for (int s = 0; s < 16; s++) {
    c += pcov[(b * 16 + s) * MM + m];
    rv += prv[(b * 16 + s) * MM + m];
  }
  c = fmaxf(c, 1e-6f);
  cov[b * MM + m] = c;
  rval[b * MM + m] = rv / c;
  rcov[b * MM + m] = fmaxf(c / support[b], 0.05f);
}

__global__ void k_part2(const float* __restrict__ attn, const float* __restrict__ rres,
                        const float* __restrict__ rval, float* __restrict__ pvar) {
  int b = blockIdx.x, sl = blockIdx.y, m = threadIdx.x;
  float val = rval[b * MM + m];
  int t0 = sl * (TT / 16);
  const float* ap = attn + (((size_t)b * TT + t0) << 8) + m;
  const float* rp = rres + b * TT + t0;
  float acc = 0.0f;
  for (int t = 0; t < TT / 16; t++) {
    float d = rp[t] - val;
    acc += ap[(size_t)t << 8] * d * d;
  }
  pvar[(b * 16 + sl) * MM + m] = acc;
}

__global__ void k_fin2(const float* __restrict__ pvar, const float* __restrict__ cov,
                       float* __restrict__ rvar) {
  int b = blockIdx.x, m = threadIdx.x;
  float s = 0.0f;
  for (int sl = 0; sl < 16; sl++) s += pvar[(b * 16 + sl) * MM + m];
  rvar[b * MM + m] = fmaxf(s / cov[b * MM + m], 1e-4f);
}

// ---------- prompt_role_fit: wave per row dot over M ----------
__global__ void __launch_bounds__(256)
k_fit(const float* __restrict__ attn, const float* __restrict__ rval,
      const float* __restrict__ mask, float* __restrict__ fit) {
  int lane = threadIdx.x & 31, wave = threadIdx.x >> 5;
  size_t row = (size_t)blockIdx.x * 8 + wave;
  int b = (int)(row >> 12);
  const float* p = attn + (row << 8);
  const float* rv = rval + b * MM;
  float s = 0.0f;
#pragma unroll
  for (int k = 0; k < 8; k++) { int m = lane + k * 32; s += p[m] * rv[m]; }
  s = wsum(s);
  if (lane == 0) fit[row] = s * mask[row];
}

extern "C" void kernel_launch(void* const* d_in, const int* in_sizes, int n_in,
                              void* d_out_v, int out_size, void* d_ws, size_t ws_size,
                              hipStream_t stream) {
  const int*   units  = (const int*)  d_in[0];
  const float* dur    = (const float*)d_in[1];
  const float* pmask  = (const float*)d_in[2];
  const float* emb    = (const float*)d_in[3];
  const float* proj_w = (const float*)d_in[4];
  const float* proj_b = (const float*)d_in[5];
  const float* c1w    = (const float*)d_in[6];
  const float* c1b    = (const float*)d_in[7];
  const float* c2w    = (const float*)d_in[8];
  const float* c2b    = (const float*)d_in[9];
  const float* lng    = (const float*)d_in[10];
  const float* lnb    = (const float*)d_in[11];
  const float* rkey   = (const float*)d_in[12];

  float* out = (float*)d_out_v;
  float* o_grate = out;              // [B,1]
  float* o_zero  = out + 16;         // [B,8]
  float* o_rval  = out + 144;        // [B,M]
  float* o_rvar  = out + 4240;       // [B,M]
  float* o_rcov  = out + 8336;       // [B,M]
  float* o_mask  = out + 12432;      // [B,T]
  float* o_ld    = out + 77968;      // [B,T]
  float* o_rres  = out + 143504;     // [B,T]
  float* o_attn  = out + 209040;     // [B,T,M]
  float* o_fit   = out + 16986256;   // [B,T]

  char* wsp = (char*)d_ws;
  size_t off = 0;
  auto alloc = [&](size_t bytes) -> void* {
    void* p = wsp + off;
    off += (bytes + 255) & ~(size_t)255;
    return p;
  };
  __bf16* hA  = (__bf16*)alloc(BTD * 2);
  __bf16* hB  = (__bf16*)alloc(BTD * 2);
  __bf16* wt1 = (__bf16*)alloc((size_t)DD * DD * KCONV * 2);
  __bf16* wt2 = (__bf16*)alloc((size_t)DD * DD * KCONV * 2);
  __bf16* rkb = (__bf16*)alloc((size_t)MM * DD * 2);
  float* cov  = (float*)alloc((size_t)BB * MM * 4);
  float* supp = (float*)alloc(256);
  float* pcov = (float*)alloc((size_t)BB * 16 * MM * 4);
  float* prv  = (float*)alloc((size_t)BB * 16 * MM * 4);
  float* pvar = (float*)alloc((size_t)BB * 16 * MM * 4);

  k_mask_logdur<<<BT / 256, 256, 0, stream>>>(dur, pmask, o_mask, o_ld);
  k_median<<<BB, 256, 0, stream>>>(o_ld, o_mask, o_grate, supp);
  k_residual<<<BT / 256, 256, 0, stream>>>(o_ld, o_mask, o_grate, o_rres);
  k_zero<<<1, 128, 0, stream>>>(o_zero, 128);
  k_prep_convw<<<(DD * DD * KCONV) / 256, 256, 0, stream>>>(c1w, wt1);
  k_prep_convw<<<(DD * DD * KCONV) / 256, 256, 0, stream>>>(c2w, wt2);
  k_prep_bf16<<<(MM * DD) / 256, 256, 0, stream>>>(rkey, rkb, MM * DD);
  k_hidden0<<<(unsigned)(BTD / 256), 256, 0, stream>>>(units, o_ld, emb, proj_w, proj_b, hA);

  dim3 gconv(TT / 256, DD / 32, BB);   // 16 x 16 x 16, 8 waves/block, 32x32 per wave
  k_conv_wmma<<<gconv, 256, 0, stream>>>(hA, wt1, c1b, hB);
  k_conv_wmma<<<gconv, 256, 0, stream>>>(hB, wt2, c2b, hA);
  k_layernorm<<<BT / 8, 256, 0, stream>>>(hA, lng, lnb, o_mask, hB);

  dim3 gsc(TT / 256, MM / 32, BB);     // 16 x 8 x 16
  k_score_wmma<<<gsc, 256, 0, stream>>>(hB, rkb, o_attn);
  k_softmax<<<BT / 8, 256, 0, stream>>>(o_attn, o_mask);

  dim3 gred(BB, 16);
  k_part1<<<gred, 256, 0, stream>>>(o_attn, o_rres, pcov, prv);
  k_fin1<<<BB, 256, 0, stream>>>(pcov, prv, supp, cov, o_rval, o_rcov);
  k_part2<<<gred, 256, 0, stream>>>(o_attn, o_rres, o_rval, pvar);
  k_fin2<<<BB, 256, 0, stream>>>(pvar, cov, o_rvar);
  k_fit<<<BT / 8, 256, 0, stream>>>(o_attn, o_rval, o_mask, o_fit);
}
